// GARNNModel_6863357739458
// MI455X (gfx1250) — matrix operand order repaired and checked by reading, not compiled, not run on hardware
//
#include <hip/hip_runtime.h>
#include <hip/hip_bf16.h>

// ---------------------------------------------------------------------------
// GARNN (graph-attention GRU encoder/decoder) for MI455X / gfx1250.
// Attention aggregation (alpha @ h) runs on WMMA f32_16x16x32_f16.
// ---------------------------------------------------------------------------

#define SEQ_LEN 12
#define HORIZON 12
#define BB      16
#define NN      512
#define DIN     2
#define DOUT    1
#define UU      64
#define NEGV    (-1.0e9f)

typedef __attribute__((ext_vector_type(16))) _Float16 v16h;
typedef __attribute__((ext_vector_type(8)))  float    v8f;

// ---------------- elementwise / small kernels ------------------------------

__global__ void mask_kernel(const float* __restrict__ adj,
                            unsigned char* __restrict__ mask, int N) {
    int idx = blockIdx.x * blockDim.x + threadIdx.x;
    if (idx >= N * N) return;
    int i = idx / N, j = idx % N;
    mask[idx] = (adj[idx] > 0.9f) || (i == j) ? 1 : 0;
}

__global__ void concat_xh_kernel(const float* __restrict__ x,
                                 const float* __restrict__ h,
                                 float* __restrict__ xh,
                                 int BN, int Din, int U) {
    int F = Din + U;
    int idx = blockIdx.x * blockDim.x + threadIdx.x;
    if (idx >= BN * F) return;
    int f = idx % F, n = idx / F;
    xh[idx] = (f < Din) ? x[n * Din + f] : h[n * U + (f - Din)];
}

__global__ void concat_xrh_kernel(const float* __restrict__ x,
                                  const float* __restrict__ r,
                                  const float* __restrict__ h,
                                  float* __restrict__ xh,
                                  int BN, int Din, int U) {
    int F = Din + U;
    int idx = blockIdx.x * blockDim.x + threadIdx.x;
    if (idx >= BN * F) return;
    int f = idx % F, n = idx / F;
    if (f < Din) xh[idx] = x[n * Din + f];
    else {
        int u = f - Din;
        xh[idx] = r[n * U + u] * h[n * U + u];
    }
}

// hfeat = xh @ W  (K = Din+U <= 66); also emit f16 copy for the WMMA path
__global__ void xw_kernel(const float* __restrict__ X,   // (BN, F)
                          const float* __restrict__ W,   // (F, G)
                          float* __restrict__ Hf,        // (BN, G)
                          _Float16* __restrict__ Hh,     // (BN, G)
                          int BN, int F, int G) {
    int idx = blockIdx.x * blockDim.x + threadIdx.x;
    if (idx >= BN * G) return;
    int g = idx % G, n = idx / G;
    const float* x = X + (size_t)n * F;
    float acc = 0.0f;
    for (int f = 0; f < F; ++f) acc += x[f] * W[f * G + g];
    Hf[idx] = acc;
    Hh[idx] = (_Float16)acc;
}

// src = h . a[0], dst = h . a[1]
__global__ void srcdst_kernel(const float* __restrict__ Hf,
                              const float* __restrict__ a,   // (2, G)
                              float* __restrict__ srcv,
                              float* __restrict__ dstv,
                              int BN, int G) {
    int n = blockIdx.x * blockDim.x + threadIdx.x;
    if (n >= BN) return;
    const float* h = Hf + (size_t)n * G;
    float s = 0.0f, d = 0.0f;
    for (int g = 0; g < G; ++g) { s += h[g] * a[g]; d += h[g] * a[G + g]; }
    srcv[n] = s;
    dstv[n] = d;
}

__global__ void sigsplit_kernel(const float* __restrict__ go,  // (BN, 2U)
                                float* __restrict__ r,
                                float* __restrict__ u,
                                int BN, int U) {
    int idx = blockIdx.x * blockDim.x + threadIdx.x;
    if (idx >= BN * U) return;
    int uu = idx % U, n = idx / U;
    float rv = go[(size_t)n * 2 * U + uu];
    float uv = go[(size_t)n * 2 * U + U + uu];
    r[idx] = 1.0f / (1.0f + __expf(-rv));
    u[idx] = 1.0f / (1.0f + __expf(-uv));
}

// h = u*h + (1-u)*tanh(c_raw)
__global__ void update_h_kernel(float* __restrict__ h,
                                const float* __restrict__ u,
                                const float* __restrict__ craw,
                                int n) {
    int idx = blockIdx.x * blockDim.x + threadIdx.x;
    if (idx >= n) return;
    float uv = u[idx];
    h[idx] = uv * h[idx] + (1.0f - uv) * tanhf(craw[idx]);
}

__global__ void proj_kernel(const float* __restrict__ h,   // (BN, U)
                            const float* __restrict__ W,   // (U, 1)
                            const float* __restrict__ b,   // (1)
                            float* __restrict__ yprev,     // (BN)
                            float* __restrict__ outslab,   // (BN)
                            int BN, int U) {
    int n = blockIdx.x * blockDim.x + threadIdx.x;
    if (n >= BN) return;
    const float* hp = h + (size_t)n * U;
    float s = 0.0f;
    for (int u = 0; u < U; ++u) s += hp[u] * W[u];
    s += b[0];
    yprev[n] = s;
    outslab[n] = s;
}

// ---------------- fused masked-softmax attention (WMMA) --------------------
//
// One wave (32 threads) handles batch b, a 16-row tile of the N x N score
// matrix, and all G output columns. Online (flash) softmax over 16x32 score
// tiles; P is built directly in 16-bit A registers in contiguous k order, V tiles
// are staged transposed in LDS so B-matrix registers are contiguous loads.
//   A order used here (16x32 f16): lane L -> row M = L&15;   elem e -> K = e + (L>>4)*16
//   B order used here (32x16 f16): lane L -> col N = L&15;   elem e -> K = e + (L>>4)*16
//   C/D layout (16x16 f32): lane L -> col N = L&15; vgpr r -> row M = r + (L>>4)*8
template <int G>
__global__ __launch_bounds__(32)
void attn_wmma_kernel(const float* __restrict__ srcv,          // (B, N)
                      const float* __restrict__ dstv,          // (B, N)
                      const _Float16* __restrict__ hf16,       // (B, N, G)
                      const unsigned char* __restrict__ mask,  // (N, N)
                      const float* __restrict__ bias,          // (G)
                      float* __restrict__ out,                 // (B, N, G)
                      int N_) {
    constexpr int GT = G / 16;
    __shared__ _Float16 vT[G * 32] __attribute__((aligned(32)));  // [g][k]

    const int lane = threadIdx.x;
    const int hi   = lane >> 4;
    const int m    = lane & 15;
    const int b    = blockIdx.x;
    const int i0   = blockIdx.y * 16;

    const int   row  = i0 + m;                 // row owned by this lane's stats
    const float srow = srcv[b * N_ + row];

    float run_max = -1.0e30f;
    float run_sum = 0.0f;
    v8f acc[GT];
#pragma unroll
    for (int t = 0; t < GT; ++t)
#pragma unroll
        for (int r = 0; r < 8; ++r) acc[t][r] = 0.0f;

    for (int jt = 0; jt < N_ / 32; ++jt) {
        const int j0 = jt * 32;

        // ---- stage V tile (32 x G) transposed into LDS ----
        __syncthreads();
        for (int c = lane; c < 32 * (G / 4); c += 32) {
            int k  = c / (G / 4);
            int g4 = (c % (G / 4)) * 4;
            const ushort4 v =
                *reinterpret_cast<const ushort4*>(hf16 + ((size_t)(b * N_ + j0 + k)) * G + g4);
            unsigned short* vs = (unsigned short*)vT;
            vs[(g4 + 0) * 32 + k] = v.x;
            vs[(g4 + 1) * 32 + k] = v.y;
            vs[(g4 + 2) * 32 + k] = v.z;
            vs[(g4 + 3) * 32 + k] = v.w;
        }
        __syncthreads();

        // ---- score tile (16 x 32) directly in A layout ----
        float ev[16];
        float lmax = -1.0e30f;
#pragma unroll
        for (int e = 0; e < 16; ++e) {
            int k = e + hi * 16;
            int j = j0 + k;
            float v = srow + dstv[b * N_ + j];
            v = (v > 0.0f) ? v : 0.2f * v;          // leaky_relu(0.2)
            v = mask[row * N_ + j] ? v : NEGV;
            ev[e]  = v;
            lmax   = fmaxf(lmax, v);
        }
        float tmax    = fmaxf(lmax, __shfl_xor(lmax, 16, 32));
        float new_max = fmaxf(run_max, tmax);
        float corr    = __expf(run_max - new_max);

        float lsum = 0.0f;
        v16h  pa;
#pragma unroll
        for (int e = 0; e < 16; ++e) {
            float p = __expf(ev[e] - new_max);
            lsum += p;
            pa[e] = (_Float16)p;
        }
        run_sum = run_sum * corr + lsum + __shfl_xor(lsum, 16, 32);
        run_max = new_max;

        // ---- rescale accumulators (row M = r + hi*8 -> stats in lane M) ----
#pragma unroll
        for (int r = 0; r < 8; ++r) {
            float cr = __shfl(corr, r + hi * 8, 32);
#pragma unroll
            for (int t = 0; t < GT; ++t) acc[t][r] *= cr;
        }

        // ---- acc += P @ V  via WMMA ----
#pragma unroll
        for (int t = 0; t < GT; ++t) {
            const _Float16* bp = &vT[(t * 16 + m) * 32 + hi * 16];
            v16h bb = *reinterpret_cast<const v16h*>(bp);
            acc[t] = __builtin_amdgcn_wmma_f32_16x16x32_f16(
                false, pa, false, bb, (short)0, acc[t], false, false);
        }
    }

    // ---- normalize, add bias, store ----
#pragma unroll
    for (int r = 0; r < 8; ++r) {
        float s   = __shfl(run_sum, r + hi * 8, 32);
        float inv = 1.0f / s;
        int   orow = i0 + r + hi * 8;
#pragma unroll
        for (int t = 0; t < GT; ++t) {
            int g = t * 16 + m;
            out[((size_t)(b * N_ + orow)) * G + g] = acc[t][r] * inv + bias[g];
        }
    }
}

// ---------------------------------------------------------------------------
// Host side
// ---------------------------------------------------------------------------

static inline int cdiv(int a, int b) { return (a + b - 1) / b; }

struct Ws {
    unsigned char* mask;
    float *h, *xh, *hfeat, *srcv, *dstv, *gat, *rb, *ub, *yprev;
    _Float16* hfeat16;
};

static void run_cell(const float* x, int Din,
                     const float* W_ru, const float* a_ru, const float* b_ru,
                     const float* W_c,  const float* a_c,  const float* b_c,
                     const Ws& w, hipStream_t stream) {
    const int BN = BB * NN;
    const int F  = Din + UU;
    const dim3 ablk(32), agrid(BB, NN / 16);

    // ru = sigmoid(GAT(concat(x,h), W_ru, a_ru) + b_ru)
    concat_xh_kernel<<<cdiv(BN * F, 256), 256, 0, stream>>>(x, w.h, w.xh, BN, Din, UU);
    xw_kernel<<<cdiv(BN * 2 * UU, 256), 256, 0, stream>>>(w.xh, W_ru, w.hfeat, w.hfeat16,
                                                          BN, F, 2 * UU);
    srcdst_kernel<<<cdiv(BN, 256), 256, 0, stream>>>(w.hfeat, a_ru, w.srcv, w.dstv, BN, 2 * UU);
    attn_wmma_kernel<2 * UU><<<agrid, ablk, 0, stream>>>(w.srcv, w.dstv, w.hfeat16, w.mask,
                                                         b_ru, w.gat, NN);
    sigsplit_kernel<<<cdiv(BN * UU, 256), 256, 0, stream>>>(w.gat, w.rb, w.ub, BN, UU);

    // c = tanh(GAT(concat(x, r*h), W_c, a_c) + b_c); h = u*h + (1-u)*c
    concat_xrh_kernel<<<cdiv(BN * F, 256), 256, 0, stream>>>(x, w.rb, w.h, w.xh, BN, Din, UU);
    xw_kernel<<<cdiv(BN * UU, 256), 256, 0, stream>>>(w.xh, W_c, w.hfeat, w.hfeat16, BN, F, UU);
    srcdst_kernel<<<cdiv(BN, 256), 256, 0, stream>>>(w.hfeat, a_c, w.srcv, w.dstv, BN, UU);
    attn_wmma_kernel<UU><<<agrid, ablk, 0, stream>>>(w.srcv, w.dstv, w.hfeat16, w.mask,
                                                     b_c, w.gat, NN);
    update_h_kernel<<<cdiv(BN * UU, 256), 256, 0, stream>>>(w.h, w.ub, w.gat, BN * UU);
}

extern "C" void kernel_launch(void* const* d_in, const int* in_sizes, int n_in,
                              void* d_out, int out_size, void* d_ws, size_t ws_size,
                              hipStream_t stream) {
    const float* inputs   = (const float*)d_in[0];
    const float* adj      = (const float*)d_in[1];
    const float* enc_W_ru = (const float*)d_in[2];
    const float* enc_a_ru = (const float*)d_in[3];
    const float* enc_b_ru = (const float*)d_in[4];
    const float* enc_W_c  = (const float*)d_in[5];
    const float* enc_a_c  = (const float*)d_in[6];
    const float* enc_b_c  = (const float*)d_in[7];
    const float* dec_W_ru = (const float*)d_in[8];
    const float* dec_a_ru = (const float*)d_in[9];
    const float* dec_b_ru = (const float*)d_in[10];
    const float* dec_W_c  = (const float*)d_in[11];
    const float* dec_a_c  = (const float*)d_in[12];
    const float* dec_b_c  = (const float*)d_in[13];
    const float* proj_W   = (const float*)d_in[14];
    const float* proj_b   = (const float*)d_in[15];
    float* out = (float*)d_out;

    const int BN = BB * NN;

    // carve workspace (256B aligned slabs)
    char* p = (char*)d_ws;
    auto carve = [&](size_t bytes) {
        void* r = (void*)p;
        p += (bytes + 255) & ~(size_t)255;
        return r;
    };
    Ws w;
    w.mask    = (unsigned char*)carve((size_t)NN * NN);
    w.h       = (float*)carve((size_t)BN * UU * 4);
    w.xh      = (float*)carve((size_t)BN * (DIN + UU) * 4);
    w.hfeat   = (float*)carve((size_t)BN * 2 * UU * 4);
    w.hfeat16 = (_Float16*)carve((size_t)BN * 2 * UU * 2);
    w.srcv    = (float*)carve((size_t)BN * 4);
    w.dstv    = (float*)carve((size_t)BN * 4);
    w.gat     = (float*)carve((size_t)BN * 2 * UU * 4);
    w.rb      = (float*)carve((size_t)BN * UU * 4);
    w.ub      = (float*)carve((size_t)BN * UU * 4);
    w.yprev   = (float*)carve((size_t)BN * 4);

    hipMemsetAsync(w.h, 0, (size_t)BN * UU * 4, stream);
    hipMemsetAsync(w.yprev, 0, (size_t)BN * 4, stream);

    mask_kernel<<<cdiv(NN * NN, 256), 256, 0, stream>>>(adj, w.mask, NN);

    // ---- encoder ----
    for (int t = 0; t < SEQ_LEN; ++t) {
        const float* x_t = inputs + (size_t)t * BN * DIN;
        run_cell(x_t, DIN, enc_W_ru, enc_a_ru, enc_b_ru, enc_W_c, enc_a_c, enc_b_c, w, stream);
    }

    // ---- decoder (autoregressive) ----
    for (int t = 0; t < HORIZON; ++t) {
        run_cell(w.yprev, DOUT, dec_W_ru, dec_a_ru, dec_b_ru, dec_W_c, dec_a_c, dec_b_c,
                 w, stream);
        proj_kernel<<<cdiv(BN, 256), 256, 0, stream>>>(w.h, proj_W, proj_b, w.yprev,
                                                       out + (size_t)t * BN, BN, UU);
    }
}
